// GMA_GCN_23510650978716
// MI455X (gfx1250) — compile-verified, matrix operand order checked by reading, not run on hardware
//
#include <hip/hip_runtime.h>
#include <hip/hip_bf16.h>

// GMA-GCN on MI455X (gfx1250): bf16 WMMA GEMMs (f32 accum) with pre-transposed
// bf16 weight staging (branch-free fragment loads), L2-resident float-atomic
// SpMM scatter, fused epilogues. All launches on `stream` (graph-capture safe).

#define NN   50000
#define EE   800000
#define KDIM 128
#define GG   5

typedef __attribute__((ext_vector_type(16))) __bf16 v16bf;
typedef __attribute__((ext_vector_type(8)))  __bf16 v8bf;
typedef __attribute__((ext_vector_type(8)))  float  v8f;

// ---------------------------------------------------------------------------
// Bt[n][k] = (n < ncols) ? B[k][n] : 0   for n in [0,ncolsP), k in [0,128)
// (weights converted to bf16 and transposed so GEMM B-fragments are two
//  contiguous 16B loads, mirroring the A-fragment pattern)
// ---------------------------------------------------------------------------
__global__ __launch_bounds__(256)
void conv_transpose_bf16(const float* __restrict__ B, __bf16* __restrict__ Bt,
                         int ncols, int ncolsP) {
  const int i = blockIdx.x * blockDim.x + threadIdx.x;
  if (i >= ncolsP * KDIM) return;
  const int n = i >> 7, k = i & (KDIM - 1);
  Bt[(size_t)n * KDIM + k] =
      (n < ncols) ? (__bf16)B[(size_t)k * ncols + n] : (__bf16)0.0f;
}

// ---------------------------------------------------------------------------
// C[M, ncols] = act(A[M,128] @ B[128,ncols] + bias)
// One wave computes a 16 x (CT*16) strip; A fragment reused across CT WMMAs.
// ACT: 0 = identity, 1 = relu, 2 = tanh.   Requires M % 16 == 0.
// ---------------------------------------------------------------------------
template<int ACT, int CT>
__global__ __launch_bounds__(256)
void gemm_bf16_wmma(const float* __restrict__ A, const __bf16* __restrict__ Bt,
                    const float* __restrict__ bias, float* __restrict__ C,
                    int M, int ncols, int ldc) {
  const int wave = threadIdx.x >> 5;
  const int lane = threadIdx.x & 31;
  const int hi   = lane >> 4;            // half-wave selects K sub-block
  const int lm   = lane & 15;
  const int row0 = ((blockIdx.x << 3) + wave) << 4;
  if (row0 >= M) return;                 // uniform per wave: EXEC stays all-1s

  v8f acc[CT] = {};
  const float* arow = A + (size_t)(row0 + lm) * KDIM;

  #pragma unroll
  for (int kk = 0; kk < KDIM; kk += 32) {
    const int kb = kk + hi * 8;
    // A fragment: lane row = row0+lm, K = kb..kb+7 and kb+16..kb+23
    const float4 a0 = *(const float4*)(arow + kb);
    const float4 a1 = *(const float4*)(arow + kb + 4);
    const float4 a2 = *(const float4*)(arow + kb + 16);
    const float4 a3 = *(const float4*)(arow + kb + 20);
    v16bf a;
    a[0] = (__bf16)a0.x;  a[1] = (__bf16)a0.y;  a[2]  = (__bf16)a0.z;  a[3]  = (__bf16)a0.w;
    a[4] = (__bf16)a1.x;  a[5] = (__bf16)a1.y;  a[6]  = (__bf16)a1.z;  a[7]  = (__bf16)a1.w;
    a[8] = (__bf16)a2.x;  a[9] = (__bf16)a2.y;  a[10] = (__bf16)a2.z;  a[11] = (__bf16)a2.w;
    a[12]= (__bf16)a3.x;  a[13]= (__bf16)a3.y;  a[14] = (__bf16)a3.z;  a[15] = (__bf16)a3.w;

    #pragma unroll
    for (int ct = 0; ct < CT; ++ct) {
      // B fragment: column n = ct*16+lm of Bt (bf16, contiguous in K)
      const __bf16* bq = Bt + (size_t)(ct * 16 + lm) * KDIM + kb;
      const v8bf blo = *(const v8bf*)(bq);
      const v8bf bhi = *(const v8bf*)(bq + 16);
      const v16bf b = __builtin_shufflevector(blo, bhi,
          0, 1, 2, 3, 4, 5, 6, 7, 8, 9, 10, 11, 12, 13, 14, 15);
      acc[ct] = __builtin_amdgcn_wmma_f32_16x16x32_bf16(
          false, a, false, b, (short)0, acc[ct], false, false);
    }
  }

  #pragma unroll
  for (int ct = 0; ct < CT; ++ct) {
    const int n = ct * 16 + lm;
    if (n >= ncols) continue;
    const float bv = bias ? bias[n] : 0.0f;
    #pragma unroll
    for (int i = 0; i < 8; ++i) {
      const int r = row0 + hi * 8 + i;   // f32 C/D layout: VGPR i -> M = i (+8)
      float v = acc[ct][i] + bv;
      if (ACT == 1) v = fmaxf(v, 0.0f);
      if (ACT == 2) v = tanhf(v);
      C[(size_t)r * ldc + n] = v;
    }
  }
}

// ---------------------------------------------------------------------------
__global__ __launch_bounds__(256)
void zero_f32(float* __restrict__ p, int n4) {
  const int i = blockIdx.x * blockDim.x + threadIdx.x;
  if (i < n4) ((float4*)p)[i] = make_float4(0.f, 0.f, 0.f, 0.f);
}

// out[dst[e]][:] += val[e] * in[src[e]][:]   -- one wave per edge, 4 f32/lane
__global__ __launch_bounds__(256)
void spmm_scatter(const int* __restrict__ src, const int* __restrict__ dst,
                  const float* __restrict__ val, const float* __restrict__ in,
                  float* __restrict__ out, int E) {
  const int e = (blockIdx.x << 3) + (threadIdx.x >> 5);
  if (e >= E) return;
  const int lane = threadIdx.x & 31;
  const int s = src[e], d = dst[e];
  const float v = val[e];
  const float4 x = ((const float4*)(in + (size_t)s * KDIM))[lane];
  float* op = out + (size_t)d * KDIM + lane * 4;
  unsafeAtomicAdd(op + 0, v * x.x);   // -> global_atomic_add_f32 (L2-resident)
  unsafeAtomicAdd(op + 1, v * x.y);
  unsafeAtomicAdd(op + 2, v * x.z);
  unsafeAtomicAdd(op + 3, v * x.w);
}

// t[i] = act(t[i] + bias[i & 127])  over [rows,128]
__global__ __launch_bounds__(256)
void bias_act(float* __restrict__ t, const float* __restrict__ bias,
              int total, int relu) {
  const int i = blockIdx.x * blockDim.x + threadIdx.x;
  if (i >= total) return;
  float v = t[i] + bias[i & (KDIM - 1)];
  if (relu) v = fmaxf(v, 0.0f);
  t[i] = v;
}

// out[r] = dot(p[r, 0:hdim], w2[0:hdim])
__global__ __launch_bounds__(256)
void att_score(const float* __restrict__ p, const float* __restrict__ w2,
               float* __restrict__ out, int rows, int hdim) {
  const int r = blockIdx.x * blockDim.x + threadIdx.x;
  if (r >= rows) return;
  const float* pr = p + (size_t)r * hdim;
  float s = 0.0f;
  for (int j = 0; j < hdim; ++j) s += pr[j] * w2[j];
  out[r] = s;
}

// softmax over K=3 branch scores, weighted sum of z[3][N][128]
__global__ __launch_bounds__(256)
void att_combine(const float* __restrict__ z, const float* __restrict__ w,
                 float* __restrict__ out, int nNodes) {
  const int i = blockIdx.x * blockDim.x + threadIdx.x;
  if (i >= nNodes * KDIM) return;
  const int nd = i >> 7;
  const float w0 = w[nd], w1 = w[nNodes + nd], w2v = w[2 * nNodes + nd];
  const float m  = fmaxf(w0, fmaxf(w1, w2v));
  const float e0 = __expf(w0 - m), e1 = __expf(w1 - m), e2 = __expf(w2v - m);
  const float inv = 1.0f / (e0 + e1 + e2);
  const size_t st = (size_t)nNodes * KDIM;
  out[i] = (e0 * z[i] + e1 * z[st + i] + e2 * z[2 * st + i]) * inv;
}

// in-place log_softmax over 10 classes per node
__global__ __launch_bounds__(256)
void log_softmax10(float* __restrict__ t, int nNodes) {
  const int nd = blockIdx.x * blockDim.x + threadIdx.x;
  if (nd >= nNodes) return;
  float* row = t + (size_t)nd * 10;
  float m = row[0];
  for (int j = 1; j < 10; ++j) m = fmaxf(m, row[j]);
  float s = 0.0f;
  for (int j = 0; j < 10; ++j) s += __expf(row[j] - m);
  const float ls = __logf(s) + m;
  for (int j = 0; j < 10; ++j) row[j] -= ls;
}

// ---------------------------------------------------------------------------
extern "C" void kernel_launch(void* const* d_in, const int* in_sizes, int n_in,
                              void* d_out, int out_size, void* d_ws, size_t ws_size,
                              hipStream_t stream) {
  const float* x    = (const float*)d_in[0];
  const int*   src  = (const int*)  d_in[1];
  const int*   dst  = (const int*)  d_in[2];
  const float* vals = (const float*)d_in[3];
  const float* W1   = (const float*)d_in[4];
  const float* b1   = (const float*)d_in[5];
  const float* W2   = (const float*)d_in[6];
  const float* b2   = (const float*)d_in[7];
  const float* aW1  = (const float*)d_in[8];
  const float* ab1  = (const float*)d_in[9];
  const float* aW2  = (const float*)d_in[10];
  const float* gW1  = (const float*)d_in[11];
  const float* gb1  = (const float*)d_in[12];
  const float* gW2  = (const float*)d_in[13];
  const float* mW   = (const float*)d_in[14];
  const float* mb   = (const float*)d_in[15];
  float* out = (float*)d_out;

  const size_t NF = (size_t)NN * KDIM;           // 6.4M floats
  // workspace layout (5*NF floats + 32KB bf16 weight staging = ~128 MB):
  float*  bufA = (float*)d_ws;                   // support / h@W2
  float*  bufB = bufA + NF;                      // spmm accum / h
  float*  zbuf = bufB + NF;                      // [3,N,128] branch outputs 0..2
  __bf16* btb  = (__bf16*)(zbuf + 3 * NF);       // transposed bf16 weights
  float*  pbuf = bufA;                           // reused after GCN loop
  float*  wbuf = bufB;                           // reused after GCN loop

  // d_out slices: (output[N,10], fadj, str_, sem, emb) -- fadj/str/sem contiguous
  float* outF = out;
  float* fadj = out + (size_t)NN * 10;
  float* strO = fadj + NF;
  float* semO = strO + NF;
  float* embO = semO + NF;

  const int gemmGridN  = (NN / 16 + 7) / 8;        // 16-row tiles, 8 waves/block
  const int gemmGridN3 = (3 * NN / 16 + 7) / 8;
  const int zeroGrid = (int)((NF / 4 + 255) / 256);
  const int nfGrid   = (int)((NF + 255) / 256);
  const int spmmGrid = EE / 8;
  const int tr128    = (128 * KDIM + 255) / 256;   // transpose grids
  const int tr16     = (16 * KDIM + 255) / 256;
  const int tr32     = (32 * KDIM + 255) / 256;

  for (int g = 0; g < GG; ++g) {
    const float* W1g = W1 + (size_t)g * KDIM * KDIM;
    const float* b1g = b1 + (size_t)g * KDIM;
    const float* W2g = W2 + (size_t)g * KDIM * KDIM;
    const float* b2g = b2 + (size_t)g * KDIM;
    const int*   sg  = src + (size_t)g * EE;
    const int*   dg  = dst + (size_t)g * EE;
    const float* vg  = vals + (size_t)g * EE;
    float* outg = (g < 3) ? (zbuf + (size_t)g * NF) : (g == 3 ? strO : semO);

    // support = x @ W1
    conv_transpose_bf16<<<tr128, 256, 0, stream>>>(W1g, btb, KDIM, KDIM);
    gemm_bf16_wmma<0, 8><<<gemmGridN, 256, 0, stream>>>(x, btb, nullptr, bufA, NN, KDIM, KDIM);
    // h = relu(spmm(support) + b1)
    zero_f32<<<zeroGrid, 256, 0, stream>>>(bufB, (int)(NF / 4));
    spmm_scatter<<<spmmGrid, 256, 0, stream>>>(sg, dg, vg, bufA, bufB, EE);
    bias_act<<<nfGrid, 256, 0, stream>>>(bufB, b1g, (int)NF, 1);
    // out = spmm(h @ W2) + b2
    conv_transpose_bf16<<<tr128, 256, 0, stream>>>(W2g, btb, KDIM, KDIM);
    gemm_bf16_wmma<0, 8><<<gemmGridN, 256, 0, stream>>>(bufB, btb, nullptr, bufA, NN, KDIM, KDIM);
    zero_f32<<<zeroGrid, 256, 0, stream>>>(outg, (int)(NF / 4));
    spmm_scatter<<<spmmGrid, 256, 0, stream>>>(sg, dg, vg, bufA, outg, EE);
    bias_act<<<nfGrid, 256, 0, stream>>>(outg, b2g, (int)NF, 0);
  }

  // attention 1: over zbuf [3N,128]
  conv_transpose_bf16<<<tr16, 256, 0, stream>>>(aW1, btb, 16, 16);
  gemm_bf16_wmma<2, 1><<<gemmGridN3, 256, 0, stream>>>(zbuf, btb, ab1, pbuf, 3 * NN, 16, 16);
  att_score<<<(3 * NN + 255) / 256, 256, 0, stream>>>(pbuf, aW2, wbuf, 3 * NN, 16);
  att_combine<<<nfGrid, 256, 0, stream>>>(zbuf, wbuf, fadj, NN);

  // attention 2: over [fadj, str_, sem] (contiguous at fadj -> flat [3N,128])
  conv_transpose_bf16<<<tr32, 256, 0, stream>>>(gW1, btb, 32, 32);
  gemm_bf16_wmma<2, 2><<<gemmGridN3, 256, 0, stream>>>(fadj, btb, gb1, pbuf, 3 * NN, 32, 32);
  att_score<<<(3 * NN + 255) / 256, 256, 0, stream>>>(pbuf, gW2, wbuf, 3 * NN, 32);
  att_combine<<<nfGrid, 256, 0, stream>>>(fadj, wbuf, embO, NN);

  // classifier: log_softmax(emb @ mW + mb)   (10 cols padded to 16 in Bt)
  conv_transpose_bf16<<<tr16, 256, 0, stream>>>(mW, btb, 10, 16);
  gemm_bf16_wmma<0, 1><<<(NN / 16 + 7) / 8, 256, 0, stream>>>(embO, btb, mb, outF, NN, 10, 10);
  log_softmax10<<<(NN + 255) / 256, 256, 0, stream>>>(outF, NN);
}
